// MambaLayer_14396730376917
// MI455X (gfx1250) — compile-verified, matrix operand order checked by reading, not run on hardware
//
#include <hip/hip_runtime.h>
#include <hip/hip_bf16.h>

// ---------------------------------------------------------------------------
// Problem constants (from reference)
// ---------------------------------------------------------------------------
#define BB 2
#define LL 2048
#define DD 1024
#define EE 2048
#define NN 16
#define DDELTA 64
#define MM (BB * LL)   // 4096 rows
#define EPS 1e-5f

// Tiled GEMM config
#define TM 128
#define TN 64
#define TK 64
#define APAD 72   // LDS row stride in halves (144B: 16B-aligned, 36-dword stride
#define BPAD 72   //  -> conflict-free stripe over the 64 LDS banks)

#define USE_ASYNC_LDS 1   // CDNA5 async global->LDS path (ASYNCcnt)

// Scan chunking (time-steps of B/C staged per barrier pair)
#define SCH 32

// ---------------------------------------------------------------------------
// Types for CDNA5 WMMA
// ---------------------------------------------------------------------------
typedef __attribute__((ext_vector_type(16))) __bf16 bf16x16;
typedef __attribute__((ext_vector_type(8)))  __bf16 bf16x8;
typedef __attribute__((ext_vector_type(8)))  float  f32x8;

__device__ __forceinline__ unsigned short f2bf(float f) {
  unsigned u = __float_as_uint(f);
  u += 0x7FFFu + ((u >> 16) & 1u);           // round-to-nearest-even
  return (unsigned short)(u >> 16);
}

__device__ __forceinline__ bf16x16 ld_frag(const unsigned short* p) {
  bf16x8 lo = *reinterpret_cast<const bf16x8*>(p);
  bf16x8 hi = *reinterpret_cast<const bf16x8*>(p + 16);
  bf16x16 f;
#pragma unroll
  for (int i = 0; i < 8; ++i) { f[i] = lo[i]; f[i + 8] = hi[i]; }
  return f;
}

// Straight-line, compile-time-specialized epilogue
template <bool HB, int ACT, bool HA, bool WF, bool WH>
__device__ __forceinline__ void epi_store(float val, int row, int col, int Nc,
                                          const float* __restrict__ bias,
                                          const float* __restrict__ addsrc,
                                          float* __restrict__ outF,
                                          unsigned short* __restrict__ outH) {
  if constexpr (HB)       val += bias[col];
  if constexpr (ACT == 1) val = (val > 20.f) ? val : log1pf(__expf(val));
  if constexpr (HA)       val += addsrc[(size_t)row * Nc + col];
  const size_t o = (size_t)row * Nc + col;
  if constexpr (WF) outF[o] = val;
  if constexpr (WH) outH[o] = f2bf(val);
}

// ---------------------------------------------------------------------------
// RMSNorm + cast to bf16 (one 256-thread block per row of D=1024)
// ---------------------------------------------------------------------------
__global__ void rmsnorm_cast_kernel(const float* __restrict__ resid,
                                    const float* __restrict__ norm_w,
                                    unsigned short* __restrict__ xbf) {
  const int row = blockIdx.x;
  const float* r = resid + (size_t)row * DD;
  float p = 0.f;
  for (int i = threadIdx.x; i < DD; i += 256) { float v = r[i]; p += v * v; }
  __shared__ float red[256];
  red[threadIdx.x] = p;
  __syncthreads();
  for (int s = 128; s > 0; s >>= 1) {
    if (threadIdx.x < s) red[threadIdx.x] += red[threadIdx.x + s];
    __syncthreads();
  }
  const float scale = rsqrtf(red[0] * (1.0f / DD) + EPS);
  for (int i = threadIdx.x; i < DD; i += 256)
    xbf[(size_t)row * DD + i] = f2bf(r[i] * scale * norm_w[i]);
}

// ---------------------------------------------------------------------------
// Elementwise casts
// ---------------------------------------------------------------------------
__global__ void cast_bf16_kernel(const float* __restrict__ src,
                                 unsigned short* __restrict__ dst, int n) {
  int i = blockIdx.x * 256 + threadIdx.x;
  if (i < n) dst[i] = f2bf(src[i]);
}

__global__ void neg_exp_kernel(const float* __restrict__ src,
                               float* __restrict__ dst, int n) {
  int i = blockIdx.x * 256 + threadIdx.x;
  if (i < n) dst[i] = -__expf(src[i]);
}

// ---------------------------------------------------------------------------
// LDS-tiled, double-buffered, async-staged WMMA bf16 GEMM.
//   out[M,Nc] = A[M,K] * W[Nc,K]^T     (M%128==0, Nc%64==0, K%64==0)
// 8 waves (4Mx2N); wave = 32x32 sub-tile = 4 f32x8 accumulators; 4 WMMA per
// 32-k substep, fragments from LDS.  Panel i+1 streams in via
// global_load_async_to_lds_b128 (ASYNCcnt) while panel i is computed.
// ---------------------------------------------------------------------------
template <bool HB, int ACT, bool HA, bool WF, bool WH>
__global__ __launch_bounds__(256)
void gemm_tiled_wmma(const unsigned short* __restrict__ Abf,
                     const unsigned short* __restrict__ Wbf,
                     const int M, const int Nc, const int K,
                     const float* __restrict__ bias,
                     const float* __restrict__ addsrc,
                     float* __restrict__ outF,
                     unsigned short* __restrict__ outH) {
  __shared__ unsigned short lA[2][TM * APAD];   // 2 x 18 KB
  __shared__ unsigned short lB[2][TN * BPAD];   // 2 x  9 KB

  const int tid  = threadIdx.x;
  const int lane = tid & 31;
  const int wid  = tid >> 5;
  const int wm   = wid & 3;          // 0..3  (M direction)
  const int wn   = wid >> 2;         // 0..1  (N direction)
  const int bm   = blockIdx.x * TM;
  const int bn   = blockIdx.y * TN;
  const int r    = lane & 15;
  const int koff = (lane >> 4) << 3; // 0 or 8

  // Per-thread copy coordinates (fixed across panels)
  const int cA0row = tid >> 3,        cA0cid = tid & 7;          // +0
  const int cB0row = tid >> 3,        cB0cid = tid & 7;          // +0
  // (A needs 4 chunks/thread: c = tid + i*256; B needs 2)

  auto stage_panel = [&](int k0, int buf) {
#pragma unroll
    for (int i = 0; i < 4; ++i) {
      const int c = tid + i * 256;            // 0..1023
      const int row = c >> 3, cid = c & 7;
      const unsigned short* g = Abf + (size_t)(bm + row) * K + k0 + cid * 8;
#if USE_ASYNC_LDS
      const unsigned lds =
          (unsigned)(uintptr_t)&lA[buf][row * APAD + cid * 8];
      asm volatile("global_load_async_to_lds_b128 %0, %1, off"
                   :: "v"(lds), "v"((unsigned long long)(uintptr_t)g)
                   : "memory");
#else
      *reinterpret_cast<bf16x8*>(&lA[buf][row * APAD + cid * 8]) =
          *reinterpret_cast<const bf16x8*>(g);
#endif
    }
#pragma unroll
    for (int i = 0; i < 2; ++i) {
      const int c = tid + i * 256;            // 0..511
      const int row = c >> 3, cid = c & 7;
      const unsigned short* g = Wbf + (size_t)(bn + row) * K + k0 + cid * 8;
#if USE_ASYNC_LDS
      const unsigned lds =
          (unsigned)(uintptr_t)&lB[buf][row * BPAD + cid * 8];
      asm volatile("global_load_async_to_lds_b128 %0, %1, off"
                   :: "v"(lds), "v"((unsigned long long)(uintptr_t)g)
                   : "memory");
#else
      *reinterpret_cast<bf16x8*>(&lB[buf][row * BPAD + cid * 8]) =
          *reinterpret_cast<const bf16x8*>(g);
#endif
    }
  };
  (void)cA0row; (void)cA0cid; (void)cB0row; (void)cB0cid;

  f32x8 acc00 = {}, acc01 = {}, acc10 = {}, acc11 = {};

  const int nk = K / TK;
  stage_panel(0, 0);
  for (int ik = 0; ik < nk; ++ik) {
    const int cur = ik & 1;
    const bool hasNext = (ik + 1) < nk;
    if (hasNext) stage_panel((ik + 1) * TK, cur ^ 1);
#if USE_ASYNC_LDS
    if (hasNext) asm volatile("s_wait_asynccnt 0x6" ::: "memory");
    else         asm volatile("s_wait_asynccnt 0x0" ::: "memory");
#endif
    __syncthreads();   // publish panel `cur` to all waves

#pragma unroll
    for (int ks = 0; ks < 2; ++ks) {
      const int kb = ks * 32 + koff;
      bf16x16 a0 = ld_frag(&lA[cur][(wm * 32 +      r) * APAD + kb]);
      bf16x16 a1 = ld_frag(&lA[cur][(wm * 32 + 16 + r) * APAD + kb]);
      bf16x16 b0 = ld_frag(&lB[cur][(wn * 32 +      r) * BPAD + kb]);
      bf16x16 b1 = ld_frag(&lB[cur][(wn * 32 + 16 + r) * BPAD + kb]);
      acc00 = __builtin_amdgcn_wmma_f32_16x16x32_bf16(false, a0, false, b0,
                                                      (short)0, acc00, false, false);
      acc01 = __builtin_amdgcn_wmma_f32_16x16x32_bf16(false, a0, false, b1,
                                                      (short)0, acc01, false, false);
      acc10 = __builtin_amdgcn_wmma_f32_16x16x32_bf16(false, a1, false, b0,
                                                      (short)0, acc10, false, false);
      acc11 = __builtin_amdgcn_wmma_f32_16x16x32_bf16(false, a1, false, b1,
                                                      (short)0, acc11, false, false);
    }
    __syncthreads();   // all waves done with `cur` before it is re-staged
  }

  const int rbase = bm + wm * 32 + ((lane >> 4) << 3);
  const int cbase = bn + wn * 32 + r;
#pragma unroll
  for (int v = 0; v < 8; ++v) {
    epi_store<HB, ACT, HA, WF, WH>(acc00[v], rbase + v,      cbase,      Nc, bias, addsrc, outF, outH);
    epi_store<HB, ACT, HA, WF, WH>(acc01[v], rbase + v,      cbase + 16, Nc, bias, addsrc, outF, outH);
    epi_store<HB, ACT, HA, WF, WH>(acc10[v], rbase + 16 + v, cbase,      Nc, bias, addsrc, outF, outH);
    epi_store<HB, ACT, HA, WF, WH>(acc11[v], rbase + 16 + v, cbase + 16, Nc, bias, addsrc, outF, outH);
  }
}

// ---------------------------------------------------------------------------
// Simple WMMA GEMM (one wave per 16x16 tile) for skinny-N cases (Nc=16/64).
// ---------------------------------------------------------------------------
template <bool WF, bool WH>
__global__ void gemm_bf16_wmma(const unsigned short* __restrict__ Abf,
                               const unsigned short* __restrict__ Wbf,
                               const int M, const int Nc, const int K,
                               float* __restrict__ outF,
                               unsigned short* __restrict__ outH) {
  const int lane = threadIdx.x & 31;
  const int wid  = (blockIdx.x * 256 + threadIdx.x) >> 5;
  const int MT   = M >> 4;
  const int m0   = (wid % MT) << 4;
  const int n0   = (wid / MT) << 4;
  const int r    = lane & 15;
  const int koff = (lane >> 4) << 3;

  const unsigned short* aRow = Abf + (size_t)(m0 + r) * K + koff;
  const unsigned short* wRow = Wbf + (size_t)(n0 + r) * K + koff;

  f32x8 acc = {};
  for (int k = 0; k < K; k += 32) {
    __builtin_prefetch(aRow + k + 256, 0, 1);
    __builtin_prefetch(wRow + k + 256, 0, 1);
    bf16x8 a_lo = *reinterpret_cast<const bf16x8*>(aRow + k);
    bf16x8 a_hi = *reinterpret_cast<const bf16x8*>(aRow + k + 16);
    bf16x8 b_lo = *reinterpret_cast<const bf16x8*>(wRow + k);
    bf16x8 b_hi = *reinterpret_cast<const bf16x8*>(wRow + k + 16);
    bf16x16 a, b;
#pragma unroll
    for (int i = 0; i < 8; ++i) {
      a[i] = a_lo[i]; a[i + 8] = a_hi[i];
      b[i] = b_lo[i]; b[i + 8] = b_hi[i];
    }
    acc = __builtin_amdgcn_wmma_f32_16x16x32_bf16(
        false, a, false, b, (short)0, acc, false, false);
  }

  const int col   = n0 + r;
  const int rbase = m0 + ((lane >> 4) << 3);
#pragma unroll
  for (int v = 0; v < 8; ++v)
    epi_store<false, 0, false, WF, WH>(acc[v], rbase + v, col, Nc,
                                       nullptr, nullptr, outF, outH);
}

// ---------------------------------------------------------------------------
// Causal depthwise conv (width 4) + bias + SiLU
// ---------------------------------------------------------------------------
__global__ void conv_silu_kernel(const float* __restrict__ xe,
                                 const float* __restrict__ conv_w,   // [E,1,4]
                                 const float* __restrict__ conv_b,   // [E]
                                 float* __restrict__ xsf,
                                 unsigned short* __restrict__ xsh) {
  const size_t idx = (size_t)blockIdx.x * 256 + threadIdx.x;
  const int e = (int)(idx % EE);
  const int l = (int)((idx / EE) % LL);
  float acc = conv_b[e];
#pragma unroll
  for (int j = 0; j < 4; ++j) {
    const int ls = l - 3 + j;
    if (ls >= 0) acc += xe[idx + (size_t)(j - 3) * EE] * conv_w[e * 4 + j];
  }
  const float s = acc / (1.f + __expf(-acc));
  xsf[idx] = s;
  xsh[idx] = f2bf(s);
}

// ---------------------------------------------------------------------------
// Sequential selective scan, B/C staged in LDS in 32-step chunks.
// grid = (B, E/256), block = 256; thread owns (b,e), h[N=16] in registers.
// ---------------------------------------------------------------------------
__global__ __launch_bounds__(256)
void scan_kernel(const float* __restrict__ delta,
                 const float* __restrict__ Bm,
                 const float* __restrict__ Cm,
                 const float* __restrict__ xsf,
                 const float* __restrict__ Aneg,
                 float* __restrict__ y) {
  const int b = blockIdx.x;
  const int e = blockIdx.y * 256 + threadIdx.x;
  __shared__ float sBC[SCH][2 * NN];   // [step][0..15]=B, [16..31]=C
  float h[NN];
  float Ae[NN];
#pragma unroll
  for (int n = 0; n < NN; ++n) { h[n] = 0.f; Ae[n] = Aneg[e * NN + n]; }

  for (int l0 = 0; l0 < LL; l0 += SCH) {
    __syncthreads();
#pragma unroll
    for (int t = 0; t < (SCH * 2 * NN) / 256; ++t) {   // 4 loads/thread
      const int c = threadIdx.x + t * 256;
      const int i = c >> 5, j = c & 31;
      const size_t row = ((size_t)b * LL + l0 + i) * NN;
      sBC[i][j] = (j < NN) ? Bm[row + j] : Cm[row + (j - NN)];
    }
    __syncthreads();
#pragma unroll 4
    for (int i = 0; i < SCH; ++i) {
      const size_t o = ((size_t)b * LL + l0 + i) * EE + e;
      const float d  = delta[o];
      const float dx = d * xsf[o];
      float acc = 0.f;
#pragma unroll
      for (int n = 0; n < NN; ++n) {
        h[n] = __expf(d * Ae[n]) * h[n] + dx * sBC[i][n];
        acc += h[n] * sBC[i][NN + n];
      }
      y[o] = acc;
    }
  }
}

// ---------------------------------------------------------------------------
// Gating epilogue:  ybf = bf16( (y + xs*W_D) * silu(skip) )
// ---------------------------------------------------------------------------
__global__ void post_scan_kernel(const float* __restrict__ y,
                                 const float* __restrict__ xsf,
                                 const float* __restrict__ skip,
                                 const float* __restrict__ W_D,
                                 unsigned short* __restrict__ ybf) {
  const size_t idx = (size_t)blockIdx.x * 256 + threadIdx.x;
  const int e = (int)(idx % EE);
  float v = y[idx] + xsf[idx] * W_D[e];
  const float s = skip[idx];
  v *= s / (1.f + __expf(-s));
  ybf[idx] = f2bf(v);
}

// ---------------------------------------------------------------------------
// Host-side launch
// ---------------------------------------------------------------------------
static inline int gemm_blocks(int M, int Nc) {
  return ((M / 16) * (Nc / 16)) / 8;
}

extern "C" void kernel_launch(void* const* d_in, const int* in_sizes, int n_in,
                              void* d_out, int out_size, void* d_ws, size_t ws_size,
                              hipStream_t stream) {
  const float* resid  = (const float*)d_in[0];
  const float* norm_w = (const float*)d_in[1];
  const float* skip_w = (const float*)d_in[2];
  const float* in_w   = (const float*)d_in[3];
  const float* conv_w = (const float*)d_in[4];
  const float* conv_b = (const float*)d_in[5];
  const float* wd1    = (const float*)d_in[6];
  const float* wd2    = (const float*)d_in[7];
  const float* wd2_b  = (const float*)d_in[8];
  const float* wb     = (const float*)d_in[9];
  const float* wc     = (const float*)d_in[10];
  const float* A_log  = (const float*)d_in[11];
  const float* W_D    = (const float*)d_in[12];
  const float* out_w  = (const float*)d_in[13];
  float* out = (float*)d_out;

  char* base = (char*)d_ws;
  size_t off = 0;
  auto alloc = [&](size_t bytes) -> void* {
    void* p = base + off;
    off += (bytes + 255) & ~(size_t)255;
    return p;
  };
  unsigned short* xbf    = (unsigned short*)alloc((size_t)MM * DD * 2);
  unsigned short* wskipb = (unsigned short*)alloc((size_t)EE * DD * 2);
  unsigned short* winb   = (unsigned short*)alloc((size_t)EE * DD * 2);
  unsigned short* woutb  = (unsigned short*)alloc((size_t)DD * EE * 2);
  unsigned short* wd1b   = (unsigned short*)alloc((size_t)DDELTA * EE * 2);
  unsigned short* wd2b   = (unsigned short*)alloc((size_t)EE * DDELTA * 2);
  unsigned short* wbb    = (unsigned short*)alloc((size_t)NN * EE * 2);
  unsigned short* wcb    = (unsigned short*)alloc((size_t)NN * EE * 2);
  float*          Aneg   = (float*)alloc((size_t)EE * NN * 4);
  float*          skipf  = (float*)alloc((size_t)MM * EE * 4);
  float*          xef    = (float*)alloc((size_t)MM * EE * 4);
  float*          xsf    = (float*)alloc((size_t)MM * EE * 4);
  unsigned short* xsh    = (unsigned short*)alloc((size_t)MM * EE * 2);
  unsigned short* d1h    = (unsigned short*)alloc((size_t)MM * DDELTA * 2);
  float*          deltaf = (float*)alloc((size_t)MM * EE * 4);
  float*          Bmf    = (float*)alloc((size_t)MM * NN * 4);
  float*          Cmf    = (float*)alloc((size_t)MM * NN * 4);
  float*          yf     = (float*)alloc((size_t)MM * EE * 4);
  unsigned short* ybf    = (unsigned short*)alloc((size_t)MM * EE * 2);

  // 1) RMSNorm + cast
  rmsnorm_cast_kernel<<<MM, 256, 0, stream>>>(resid, norm_w, xbf);

  // 2) weight casts + A = -exp(A_log)
  cast_bf16_kernel<<<(EE * DD) / 256, 256, 0, stream>>>(skip_w, wskipb, EE * DD);
  cast_bf16_kernel<<<(EE * DD) / 256, 256, 0, stream>>>(in_w,   winb,   EE * DD);
  cast_bf16_kernel<<<(DD * EE) / 256, 256, 0, stream>>>(out_w,  woutb,  DD * EE);
  cast_bf16_kernel<<<(DDELTA * EE) / 256, 256, 0, stream>>>(wd1, wd1b, DDELTA * EE);
  cast_bf16_kernel<<<(EE * DDELTA) / 256, 256, 0, stream>>>(wd2, wd2b, EE * DDELTA);
  cast_bf16_kernel<<<(NN * EE) / 256, 256, 0, stream>>>(wb, wbb, NN * EE);
  cast_bf16_kernel<<<(NN * EE) / 256, 256, 0, stream>>>(wc, wcb, NN * EE);
  neg_exp_kernel<<<(EE * NN) / 256, 256, 0, stream>>>(A_log, Aneg, EE * NN);

  // 3) skip = x @ skip_w^T ; xe = x @ in_w^T   (tiled, 4096x2048, K=1024)
  gemm_tiled_wmma<false, 0, false, true, false>
      <<<dim3(MM / TM, EE / TN), 256, 0, stream>>>(
          xbf, wskipb, MM, EE, DD, nullptr, nullptr, skipf, nullptr);
  gemm_tiled_wmma<false, 0, false, true, false>
      <<<dim3(MM / TM, EE / TN), 256, 0, stream>>>(
          xbf, winb, MM, EE, DD, nullptr, nullptr, xef, nullptr);

  // 4) depthwise causal conv + SiLU
  conv_silu_kernel<<<(MM * EE) / 256, 256, 0, stream>>>(xef, conv_w, conv_b, xsf, xsh);

  // 5) delta = softplus((xs @ wd1^T) @ wd2^T + b)
  gemm_bf16_wmma<false, true><<<gemm_blocks(MM, DDELTA), 256, 0, stream>>>(
      xsh, wd1b, MM, DDELTA, EE, nullptr, d1h);
  gemm_tiled_wmma<true, 1, false, true, false>
      <<<dim3(MM / TM, EE / TN), 256, 0, stream>>>(
          d1h, wd2b, MM, EE, DDELTA, wd2_b, nullptr, deltaf, nullptr);

  // 6) Bm = xs @ wb^T ; Cm = xs @ wc^T   (skinny: Nc=16, K=2048)
  gemm_bf16_wmma<true, false><<<gemm_blocks(MM, NN), 256, 0, stream>>>(
      xsh, wbb, MM, NN, EE, Bmf, nullptr);
  gemm_bf16_wmma<true, false><<<gemm_blocks(MM, NN), 256, 0, stream>>>(
      xsh, wcb, MM, NN, EE, Cmf, nullptr);

  // 7) sequential selective scan (chunked B/C staging)
  scan_kernel<<<dim3(BB, EE / 256), 256, 0, stream>>>(deltaf, Bmf, Cmf, xsf, Aneg, yf);

  // 8) gating epilogue -> bf16
  post_scan_kernel<<<(MM * EE) / 256, 256, 0, stream>>>(yf, xsf, skipf, W_D, ybf);

  // 9) out = resid + y @ out_w^T   (tiled, 4096x1024, K=2048)
  gemm_tiled_wmma<false, 0, true, true, false>
      <<<dim3(MM / TM, DD / TN), 256, 0, stream>>>(
          ybf, woutb, MM, DD, EE, nullptr, resid, out, nullptr);
}
